// RLSTMCell_31550829757136
// MI455X (gfx1250) — matrix operand order repaired and checked by reading, not compiled, not run on hardware
//
#include <hip/hip_runtime.h>
#include <hip/hip_bf16.h>
#include <math.h>

// ---------------------------------------------------------------------------
// RLSTM cell for MI455X (gfx1250, wave32).
// All seven GEMMs (K=512) run through a bf16 WMMA GEMM
// (v_wmma_f32_16x16x32_bf16, f32 accumulate). Attention / softmax / layernorm
// / gates are f32 VALU epilogue kernels (<1% of FLOPs).
// Straight (non-transposing) global->LDS staging uses CDNA5 async-to-LDS
// (global_load_async_to_lds_b128, ASYNCcnt) when the toolchain exposes it.
// ---------------------------------------------------------------------------

typedef __bf16 bf16_t;
typedef __attribute__((ext_vector_type(16))) __bf16      v16bf;
typedef __attribute__((ext_vector_type(8)))  __bf16      v8bf;
typedef __attribute__((ext_vector_type(8)))  float       v8f;
typedef __attribute__((ext_vector_type(8)))  unsigned int u32x8;
typedef int v4i __attribute__((vector_size(16)));         // matches builtin sig

#define DD   512                      // MEMDIM
#define SS   16                       // MEMSIZE
#define BB   4096                     // BATCH
#define MS   (BB * SS)                // 65536 rows (b,s)
#define MM   (BB * (SS + 1))          // 69632 rows (b, s|x)
#define TD   (3 * DD)                 // 1536
#define NOUT ((long)BB * SS * DD)     // 33,554,432 elements per output tensor

// ---- CDNA5 async global->LDS copy (guarded: falls back to sync copy) -------
#if __has_builtin(__builtin_amdgcn_global_load_async_to_lds_b128)
#define ASYNC_LDS 1
#else
#define ASYNC_LDS 0
#endif

__device__ __forceinline__ void cp16_lds(bf16_t* lds_dst, const bf16_t* gsrc) {
#if ASYNC_LDS
  // signature (from compiler diagnostic): (v4i AS1*, v4i AS3*, imm off, imm cpol)
  __builtin_amdgcn_global_load_async_to_lds_b128(
      (__attribute__((address_space(1))) v4i*)(void*)(const_cast<bf16_t*>(gsrc)),
      (__attribute__((address_space(3))) v4i*)(void*)(lds_dst), 0, 0);
#else
  *(uint4*)lds_dst = *(const uint4*)gsrc;
#endif
}

__device__ __forceinline__ void async_join() {
#if ASYNC_LDS
#if __has_builtin(__builtin_amdgcn_s_wait_asynccnt)
  __builtin_amdgcn_s_wait_asynccnt(0);
#else
  asm volatile("s_wait_asynccnt 0" ::: "memory");
#endif
#endif
}

// ------------------------------- elementwise --------------------------------

__global__ void cvt_f32_bf16(const float* __restrict__ src,
                             bf16_t* __restrict__ dst, long n) {
  long i = (long)blockIdx.x * blockDim.x + threadIdx.x;
  long st = (long)gridDim.x * blockDim.x;
  for (; i < n; i += st) dst[i] = (bf16_t)src[i];
}

// memb layout: (B, S+1, D) bf16 ; rows 0..15 = c slots, row 16 = input
__global__ void build_mem(const float* __restrict__ c,
                          const float* __restrict__ x,
                          bf16_t* __restrict__ memb) {
  long n = (long)MM * DD;
  long i = (long)blockIdx.x * blockDim.x + threadIdx.x;
  long st = (long)gridDim.x * blockDim.x;
  for (; i < n; i += st) {
    int  d    = (int)(i & (DD - 1));
    long r    = i >> 9;
    int  slot = (int)(r % 17);
    long b    = r / 17;
    float v = (slot < SS) ? c[(b * SS + slot) * DD + d] : x[b * DD + d];
    memb[i] = (bf16_t)v;
  }
}

// ------------------------------- WMMA GEMM ----------------------------------
// C(MxN) = A(MxK,bf16,row-major) * B(KxN,bf16,row-major), K multiple of 32,
// M,N multiples of 128. lda=K, ldb=N, ldc=N.
// EPI:   0 -> f32 store, 1 -> bf16 store, 2 -> bf16 sigmoid(acc+xz+bias)
// AMODE: 0 -> row m, 1 -> row m+(m>>4) (c-slots in memb), 2 -> row m*17+16 (x)

template <int EPI, int AMODE>
__global__ __launch_bounds__(256) void gemm_bf16(
    const bf16_t* __restrict__ A, const bf16_t* __restrict__ B,
    float* __restrict__ Cf, bf16_t* __restrict__ Cb,
    int M, int N, int K,
    const float* __restrict__ xz, const float* __restrict__ bias) {
  constexpr int SK = 40;                       // padded LDS row stride (bf16)
  __shared__ bf16_t As[2][128 * SK];
  __shared__ bf16_t Bs[2][128 * SK];

  const int tid = threadIdx.x;
  const int bm0 = blockIdx.y * 128;
  const int bn0 = blockIdx.x * 128;
  const int NK  = K >> 5;

  // ---- global loaders ----
  const int ar = tid >> 1;                     // tile row 0..127
  const int ak = (tid & 1) * 16;               // 0 / 16
  long a_row = bm0 + ar;
  if (AMODE == 1)      a_row = a_row + (a_row >> 4);
  else if (AMODE == 2) a_row = a_row * 17 + 16;
  const bf16_t* Ag = A + a_row * (long)K + ak;

  const int bk = tid & 31;                     // k within tile
  const int bn = (tid >> 5) * 16;              // n chunk base
  const bf16_t* Bg = B + (long)bk * N + (bn0 + bn);

  uint4 arg0, arg1, brg0, brg1;

  auto stageA_async = [&](int buf, int kb) {   // A tile: no transpose -> async
    bf16_t* dst = &As[buf][ar * SK + ak];
    const bf16_t* src = Ag + (long)kb * 32;
    cp16_lds(dst, src);
    cp16_lds(dst + 8, src + 8);
  };
  auto gloadA = [&](int kb) {
    const uint4* pa = (const uint4*)(Ag + (long)kb * 32);
    arg0 = pa[0]; arg1 = pa[1];
  };
  auto sstoreA = [&](int buf) {
    uint4* pa = (uint4*)(&As[buf][ar * SK + ak]);   // 16B aligned (80B rows)
    pa[0] = arg0; pa[1] = arg1;
  };
  auto gloadB = [&](int kb) {
    const uint4* pb = (const uint4*)(Bg + (long)kb * 32 * N);
    brg0 = pb[0]; brg1 = pb[1];
  };
  auto sstoreB = [&](int buf) {                // B tile: transposed scatter
    const bf16_t* v0 = (const bf16_t*)&brg0;
    const bf16_t* v1 = (const bf16_t*)&brg1;
#pragma unroll
    for (int j = 0; j < 8; ++j) Bs[buf][(bn + j) * SK + bk] = v0[j];
#pragma unroll
    for (int j = 0; j < 8; ++j) Bs[buf][(bn + 8 + j) * SK + bk] = v1[j];
  };

  // ---- per-wave tiling: 8 waves -> 2x4 grid of 64x32 wave tiles ----
  const int wave = tid >> 5;
  const int lane = tid & 31;
  const int wm = (wave >> 2) * 64;
  const int wn = (wave & 3) * 32;
  const int lm = lane & 15;
  const int lh = lane >> 4;

  v8f acc[4][2];
  const v8f vzero = {0.f, 0.f, 0.f, 0.f, 0.f, 0.f, 0.f, 0.f};
#pragma unroll
  for (int mi = 0; mi < 4; ++mi)
#pragma unroll
    for (int ni = 0; ni < 2; ++ni) acc[mi][ni] = vzero;

  auto compute = [&](int buf) {
    const unsigned int* A32 = (const unsigned int*)As[buf];
    const unsigned int* B32 = (const unsigned int*)Bs[buf];
    v16bf afr[4], bfr[2];
#pragma unroll
    for (int mi = 0; mi < 4; ++mi) {
      const unsigned int* p = A32 + (wm + mi * 16 + lm) * (SK / 2);
      u32x8 t;
#pragma unroll
      for (int v = 0; v < 8; ++v) t[v] = p[(v & 3) + 4 * lh + 8 * (v >> 2)];
      afr[mi] = __builtin_bit_cast(v16bf, t);
    }
#pragma unroll
    for (int ni = 0; ni < 2; ++ni) {
      const unsigned int* p = B32 + (wn + ni * 16 + lm) * (SK / 2);
      u32x8 t;
#pragma unroll
      for (int v = 0; v < 8; ++v) t[v] = p[(v & 3) + 4 * lh + 8 * (v >> 2)];
      bfr[ni] = __builtin_bit_cast(v16bf, t);
    }
#pragma unroll
    for (int mi = 0; mi < 4; ++mi)
#pragma unroll
      for (int ni = 0; ni < 2; ++ni)
        acc[mi][ni] = __builtin_amdgcn_wmma_f32_16x16x32_bf16(
            false, afr[mi], false, bfr[ni], (short)0, acc[mi][ni], false, false);
  };

  // ---- pipelined K loop (double-buffered LDS) ----
#if ASYNC_LDS
  stageA_async(0, 0);
  gloadB(0);
  sstoreB(0);
  async_join();
  __syncthreads();
  for (int kb = 0; kb < NK; ++kb) {
    if (kb + 1 < NK) {
      stageA_async((kb + 1) & 1, kb + 1);      // DMA overlaps with compute
      gloadB(kb + 1);
    }
    if (kb + 2 < NK) {
      __builtin_prefetch(Ag + (long)(kb + 2) * 32, 0, 1);
      __builtin_prefetch(Bg + (long)(kb + 2) * 32 * N, 0, 1);
    }
    compute(kb & 1);
    if (kb + 1 < NK) {
      sstoreB((kb + 1) & 1);
      async_join();
    }
    __syncthreads();
  }
#else
  gloadA(0);
  gloadB(0);
  sstoreA(0);
  sstoreB(0);
  __syncthreads();
  for (int kb = 0; kb < NK; ++kb) {
    if (kb + 1 < NK) { gloadA(kb + 1); gloadB(kb + 1); }
    if (kb + 2 < NK) {
      __builtin_prefetch(Ag + (long)(kb + 2) * 32, 0, 1);
      __builtin_prefetch(Bg + (long)(kb + 2) * 32 * N, 0, 1);
    }
    compute(kb & 1);
    if (kb + 1 < NK) { sstoreA((kb + 1) & 1); sstoreB((kb + 1) & 1); }
    __syncthreads();
  }
#endif

  // ---- epilogue ----
#pragma unroll
  for (int mi = 0; mi < 4; ++mi)
#pragma unroll
    for (int ni = 0; ni < 2; ++ni)
#pragma unroll
      for (int r = 0; r < 8; ++r) {
        long gm = bm0 + wm + mi * 16 + r + 8 * lh;
        long gn = bn0 + wn + ni * 16 + lm;
        float val = acc[mi][ni][r];
        if (EPI == 0) {
          Cf[gm * N + gn] = val;
        } else if (EPI == 1) {
          Cb[gm * N + gn] = (bf16_t)val;
        } else {  // gate: sigmoid(hz + xz + bias)
          float z = val + xz[(gm >> 4) * N + gn] + bias[gn];
          Cb[gm * N + gn] = (bf16_t)(1.0f / (1.0f + expf(-z)));
        }
      }
}

// ------------------------------ attention -----------------------------------
// One block per batch element. scores = sqrt(D) * Q K^T (16x17, K=512),
// softmax over 17, c_upd = att @ V. Vectorized (b128) LDS traffic.

__global__ __launch_bounds__(256) void attn_kernel(
    const bf16_t* __restrict__ Qb, const bf16_t* __restrict__ Kb,
    const bf16_t* __restrict__ Vb, bf16_t* __restrict__ out) {
  __shared__ bf16_t Qs[16 * DD];
  __shared__ bf16_t Ks[17 * DD];
  __shared__ bf16_t Vs[17 * DD];
  __shared__ float  att[16 * 17];

  const int b = blockIdx.x, tid = threadIdx.x;

  const bf16_t* gq = Qb + (long)b * 16 * DD;
  const bf16_t* gk = Kb + (long)b * 17 * DD;
  const bf16_t* gv = Vb + (long)b * 17 * DD;
  for (int i = tid; i < 16 * DD / 8; i += 256) cp16_lds(Qs + i * 8, gq + i * 8);
  for (int i = tid; i < 17 * DD / 8; i += 256) cp16_lds(Ks + i * 8, gk + i * 8);
  for (int i = tid; i < 17 * DD / 8; i += 256) cp16_lds(Vs + i * 8, gv + i * 8);
  async_join();
  __syncthreads();

  for (int p = tid; p < 16 * 17; p += 256) {
    int j = p / 17, l = p % 17;
    const uint4* qp = (const uint4*)(Qs + j * DD);
    const uint4* kp = (const uint4*)(Ks + l * DD);
    float s = 0.f;
    for (int d8 = 0; d8 < DD / 8; ++d8) {
      v8bf qa = __builtin_bit_cast(v8bf, qp[d8]);
      v8bf ka = __builtin_bit_cast(v8bf, kp[d8]);
#pragma unroll
      for (int e = 0; e < 8; ++e) s += (float)qa[e] * (float)ka[e];
    }
    att[p] = s * 22.627416997969522f;  // * sqrt(512)
  }
  __syncthreads();

  if (tid < 16) {
    float mx = -1e30f;
    for (int l = 0; l < 17; ++l) mx = fmaxf(mx, att[tid * 17 + l]);
    float sum = 0.f;
    for (int l = 0; l < 17; ++l) {
      float e = expf(att[tid * 17 + l] - mx);
      att[tid * 17 + l] = e;
      sum += e;
    }
    float inv = 1.f / sum;
    for (int l = 0; l < 17; ++l) att[tid * 17 + l] *= inv;
  }
  __syncthreads();

  // c_upd = att @ V : each thread produces 8 consecutive d's (b128 IO)
  for (int q = tid; q < 16 * DD / 8; q += 256) {
    int j  = q >> 6;            // 64 8-wide chunks per slot row
    int d8 = q & 63;
    float acc[8] = {0.f, 0.f, 0.f, 0.f, 0.f, 0.f, 0.f, 0.f};
#pragma unroll 1
    for (int l = 0; l < 17; ++l) {
      v8bf va = __builtin_bit_cast(v8bf, ((const uint4*)(Vs + l * DD))[d8]);
      float a = att[j * 17 + l];
#pragma unroll
      for (int e = 0; e < 8; ++e) acc[e] += a * (float)va[e];
    }
    v8bf ob;
#pragma unroll
    for (int e = 0; e < 8; ++e) ob[e] = (bf16_t)acc[e];
    *(uint4*)(out + ((long)b * 16 + j) * DD + d8 * 8) =
        __builtin_bit_cast(uint4, ob);
  }
}

// ------------------------------ layernorm -----------------------------------
// One block per row of 512. MODE 0: y=relu(ln(x)) -> bf16 ; MODE 1: ln -> f32.

template <int MODE>
__global__ __launch_bounds__(256) void ln_kernel(
    const float* __restrict__ X, const float* __restrict__ gain,
    const float* __restrict__ bias, bf16_t* __restrict__ outb,
    float* __restrict__ outf) {
  __shared__ float r1[256], r2[256];
  __shared__ float mu_s, inv_s;
  const int tid = threadIdx.x;
  const long row = blockIdx.x;
  const float* x = X + row * DD;
  float x0 = x[tid], x1 = x[tid + 256];
  r1[tid] = x0 + x1;
  r2[tid] = x0 * x0 + x1 * x1;
  __syncthreads();
  for (int s = 128; s > 0; s >>= 1) {
    if (tid < s) { r1[tid] += r1[tid + s]; r2[tid] += r2[tid + s]; }
    __syncthreads();
  }
  if (tid == 0) {
    float mu  = r1[0] * (1.f / DD);
    float var = fmaxf(r2[0] * (1.f / DD) - mu * mu, 0.f);
    mu_s  = mu;
    inv_s = 1.f / (sqrtf(var) + 1e-9f);
  }
  __syncthreads();
  float mu = mu_s, inv = inv_s;
#pragma unroll
  for (int e = 0; e < 2; ++e) {
    int d = tid + e * 256;
    float xv = (e == 0) ? x0 : x1;
    float y = gain[d] * (xv - mu) * inv + bias[d];
    if (MODE == 0) outb[row * DD + d] = (bf16_t)fmaxf(y, 0.f);
    else           outf[row * DD + d] = y;
  }
}

// ------------------------------- gate fuse ----------------------------------
// c_new = f*c_old + i*tanh(cu) ; h_new = o*tanh(c_new). out = [h_new | c_new].

__global__ void gate_kernel(const bf16_t* __restrict__ zsig,
                            const float* __restrict__ c_old,
                            const float* __restrict__ cu,
                            float* __restrict__ out) {
  long i = (long)blockIdx.x * blockDim.x + threadIdx.x;
  long st = (long)gridDim.x * blockDim.x;
  for (; i < NOUT; i += st) {
    long m = i >> 9;
    int  d = (int)(i & (DD - 1));
    const bf16_t* zr = zsig + m * TD;
    float ig = (float)zr[d];
    float fg = (float)zr[DD + d];
    float og = (float)zr[2 * DD + d];
    float cn = fg * c_old[i] + ig * tanhf(cu[i]);
    out[i]        = og * tanhf(cn);
    out[NOUT + i] = cn;
  }
}

// ------------------------------- launcher -----------------------------------

extern "C" void kernel_launch(void* const* d_in, const int* in_sizes, int n_in,
                              void* d_out, int out_size, void* d_ws,
                              size_t ws_size, hipStream_t stream) {
  const float* x    = (const float*)d_in[0];
  const float* h0   = (const float*)d_in[1];
  const float* c0   = (const float*)d_in[2];
  const float* kern = (const float*)d_in[3];
  const float* rker = (const float*)d_in[4];
  const float* Wq   = (const float*)d_in[5];
  const float* Wk   = (const float*)d_in[6];
  const float* Wv   = (const float*)d_in[7];
  const float* m1   = (const float*)d_in[8];
  const float* m2   = (const float*)d_in[9];
  const float* g1   = (const float*)d_in[10];
  const float* g2   = (const float*)d_in[11];
  const float* b1   = (const float*)d_in[12];
  const float* b2   = (const float*)d_in[13];
  const float* bias = (const float*)d_in[14];
  float* out = (float*)d_out;

  char* ws = (char*)d_ws;
  size_t off = 0;
  auto alloc = [&](size_t bytes) -> void* {
    off = (off + 255) & ~(size_t)255;
    void* p = ws + off;
    off += bytes;
    return p;
  };

  bf16_t* hb    = (bf16_t*)alloc((size_t)MS * DD * 2);
  bf16_t* memb  = (bf16_t*)alloc((size_t)MM * DD * 2);
  bf16_t* wqb   = (bf16_t*)alloc((size_t)DD * DD * 2);
  bf16_t* wkb   = (bf16_t*)alloc((size_t)DD * DD * 2);
  bf16_t* wvb   = (bf16_t*)alloc((size_t)DD * DD * 2);
  bf16_t* m1b   = (bf16_t*)alloc((size_t)DD * DD * 2);
  bf16_t* m2b   = (bf16_t*)alloc((size_t)DD * DD * 2);
  bf16_t* kernb = (bf16_t*)alloc((size_t)DD * TD * 2);
  bf16_t* rkerb = (bf16_t*)alloc((size_t)DD * TD * 2);
  bf16_t* Qb    = (bf16_t*)alloc((size_t)MS * DD * 2);
  bf16_t* Kmb   = (bf16_t*)alloc((size_t)MM * DD * 2);
  bf16_t* Vb    = (bf16_t*)alloc((size_t)MM * DD * 2);
  bf16_t* cub   = (bf16_t*)alloc((size_t)MS * DD * 2);
  bf16_t* u1b   = (bf16_t*)alloc((size_t)MS * DD * 2);
  bf16_t* zsig  = (bf16_t*)alloc((size_t)MS * TD * 2);
  float*  gbuf  = (float*)alloc((size_t)MS * DD * 4);
  float*  cu2   = (float*)alloc((size_t)MS * DD * 4);
  float*  xzf   = (float*)alloc((size_t)BB * TD * 4);

  const int T = 256;
  auto blocks = [](long n) { return (unsigned)((n + 255) / 256); };

  // bf16 conversions / packing
  cvt_f32_bf16<<<blocks((long)MS * DD), T, 0, stream>>>(h0, hb, (long)MS * DD);
  build_mem<<<blocks((long)MM * DD), T, 0, stream>>>(c0, x, memb);
  cvt_f32_bf16<<<blocks((long)DD * DD), T, 0, stream>>>(Wq, wqb, (long)DD * DD);
  cvt_f32_bf16<<<blocks((long)DD * DD), T, 0, stream>>>(Wk, wkb, (long)DD * DD);
  cvt_f32_bf16<<<blocks((long)DD * DD), T, 0, stream>>>(Wv, wvb, (long)DD * DD);
  cvt_f32_bf16<<<blocks((long)DD * DD), T, 0, stream>>>(m1, m1b, (long)DD * DD);
  cvt_f32_bf16<<<blocks((long)DD * DD), T, 0, stream>>>(m2, m2b, (long)DD * DD);
  cvt_f32_bf16<<<blocks((long)DD * TD), T, 0, stream>>>(kern, kernb, (long)DD * TD);
  cvt_f32_bf16<<<blocks((long)DD * TD), T, 0, stream>>>(rker, rkerb, (long)DD * TD);

  // Q / K / V projections (bf16 out)
  gemm_bf16<1, 1><<<dim3(DD / 128, MS / 128), T, 0, stream>>>(
      memb, wqb, nullptr, Qb, MS, DD, DD, nullptr, nullptr);
  gemm_bf16<1, 0><<<dim3(DD / 128, MM / 128), T, 0, stream>>>(
      memb, wkb, nullptr, Kmb, MM, DD, DD, nullptr, nullptr);
  gemm_bf16<1, 0><<<dim3(DD / 128, MM / 128), T, 0, stream>>>(
      memb, wvb, nullptr, Vb, MM, DD, DD, nullptr, nullptr);

  // attention over 17 slots
  attn_kernel<<<BB, T, 0, stream>>>(Qb, Kmb, Vb, cub);

  // MLP with layernorms
  gemm_bf16<0, 0><<<dim3(DD / 128, MS / 128), T, 0, stream>>>(
      cub, m1b, gbuf, nullptr, MS, DD, DD, nullptr, nullptr);
  ln_kernel<0><<<MS, T, 0, stream>>>(gbuf, g1, b1, u1b, nullptr);
  gemm_bf16<0, 0><<<dim3(DD / 128, MS / 128), T, 0, stream>>>(
      u1b, m2b, gbuf, nullptr, MS, DD, DD, nullptr, nullptr);
  ln_kernel<1><<<MS, T, 0, stream>>>(gbuf, g2, b2, nullptr, cu2);

  // gates: xz (f32), then hz GEMM with fused sigmoid(hz + xz + bias) -> bf16
  gemm_bf16<0, 2><<<dim3(TD / 128, BB / 128), T, 0, stream>>>(
      memb, kernb, xzf, nullptr, BB, TD, DD, nullptr, nullptr);
  gemm_bf16<2, 0><<<dim3(TD / 128, MS / 128), T, 0, stream>>>(
      hb, rkerb, nullptr, zsig, MS, TD, DD, xzf, bias);

  // final elementwise fuse -> [h_new | c_new]
  gate_kernel<<<blocks(NOUT), T, 0, stream>>>(zsig, c0, cu2, out);
}
